// GCNeXtC_31430570672674
// MI455X (gfx1250) — matrix operand order repaired and checked by reading, not compiled, not run on hardware
//
#include <hip/hip_runtime.h>
#include <hip/hip_bf16.h>
#include <stdint.h>

// ---------------------------------------------------------------------------
// Problem constants (from reference): B=16, C=256, N=2048, K=10, G=32, W=128
// ---------------------------------------------------------------------------
#define BB 16
#define CC 256
#define NN 2048
#define KK 10
#define WW 128

typedef __bf16 bf16_t;
typedef __attribute__((ext_vector_type(16))) __bf16 v16bf;
typedef __attribute__((ext_vector_type(8)))  __bf16 v8bf;
typedef __attribute__((ext_vector_type(8)))  float  v8f;

union BF16Frag { v16bf v; v8bf h[2]; };

// ---------------------------------------------------------------------------
// Workspace layout (bytes, all 256-aligned). Total ~210 MB.
// ---------------------------------------------------------------------------
static const size_t OFF_XC16 = 0;          // (B,C,N) bf16      16777216
static const size_t OFF_XT16 = 16777216;   // (B,N,C) bf16      16777216
static const size_t OFF_T1   = 33554432;   // (B,W,N) bf16       8388608
static const size_t OFF_T2   = 41943040;   // (B,W,N) bf16       8388608
static const size_t OFF_ZT   = 50331648;   // (B,N,2W) bf16     16777216
static const size_t OFF_WBUF = 67108864;   // (B,W,K,N) bf16    83886080
static const size_t OFF_TOUT = 150994944;  // (B,C,N) f32       33554432
static const size_t OFF_SOUT = 184549376;  // (B,C,N) f32       33554432
static const size_t OFF_XX   = 218103808;  // (B,N)   f32         131072
static const size_t OFF_IDX  = 218234880;  // (B,N,K) i32        1310720
static const size_t OFF_TW1B = 219545600;  // (W,C)   bf16         65536
static const size_t OFF_TW3B = 219611136;  // (C,W)   bf16         65536
static const size_t OFF_SW1S = 219676672;  // (2W,C)  bf16 stacked 131072
static const size_t OFF_SW3B = 219807744;  // (C,W)   bf16         65536

// ---------------------------------------------------------------------------
// WMMA fragment helpers (CDNA5 V_WMMA_F32_16X16X32_BF16 layouts, ISA 7.12.2):
//  A 16x32 bf16: lane L holds row M=L&15; halfwords 0-7 = K=(L>>4)*8+0..7,
//                halfwords 8-15 = K=16+(L>>4)*8+0..7  -> two 16B runs.
//  B 32x16 bf16: lane L holds column N=L&15; halfwords 0-7 = K=(L>>4)*8+0..7,
//                halfwords 8-15 = K=16+(L>>4)*8+0..7 (strided gather).
//  C/D 16x16 f32: lane L holds column N=L&15; VGPR r = row M=r+8*(L>>4).
// ---------------------------------------------------------------------------
__device__ __forceinline__ v16bf load_a_frag(const bf16_t* __restrict__ base,
                                             int row0, int ld, int k0, int lane) {
    const int lo = lane & 15, hi = lane >> 4;
    const bf16_t* p = base + (size_t)(row0 + lo) * ld + k0 + hi * 8;
    BF16Frag f;
    f.h[0] = *(const v8bf*)(p);
    f.h[1] = *(const v8bf*)(p + 16);
    return f.v;
}

__device__ __forceinline__ v16bf load_b_frag(const bf16_t* __restrict__ base,
                                             int k0, size_t ld, int col0, int lane) {
    const int lo = lane & 15, hi = lane >> 4;
    const bf16_t* p = base + (size_t)(k0 + hi * 8) * ld + col0 + lo;
    v16bf v;
    #pragma unroll
    for (int s = 0; s < 8; ++s) {
        v[s]     = p[(size_t)s * ld];
        v[s + 8] = p[(size_t)(s + 16) * ld];
    }
    return v;
}

__device__ __forceinline__ v8f wmma_bf16(v16bf a, v16bf b, v8f c) {
    return __builtin_amdgcn_wmma_f32_16x16x32_bf16(false, a, false, b,
                                                   (short)0, c, false, false);
}

__device__ __forceinline__ void topk_insert(float v, int id, float (&tv)[KK], int (&ti)[KK]) {
    if (v > tv[KK - 1]) {
        tv[KK - 1] = v; ti[KK - 1] = id;
        #pragma unroll
        for (int s = KK - 1; s > 0; --s)
            if (tv[s] > tv[s - 1]) {
                float tf = tv[s]; tv[s] = tv[s - 1]; tv[s - 1] = tf;
                int   tq = ti[s]; ti[s] = ti[s - 1]; ti[s - 1] = tq;
            }
    }
}

// ---------------------------------------------------------------------------
// Prep kernels
// ---------------------------------------------------------------------------
__global__ void convw_kernel(const float* __restrict__ tw1, const float* __restrict__ tw3,
                             const float* __restrict__ sw1, const float* __restrict__ sw3,
                             bf16_t* __restrict__ tw1b, bf16_t* __restrict__ tw3b,
                             bf16_t* __restrict__ sw1s, bf16_t* __restrict__ sw3b) {
    int id = blockIdx.x * blockDim.x + threadIdx.x;       // grid covers 65536
    if (id < WW * CC) tw1b[id] = (bf16_t)tw1[id];
    if (id < CC * WW) tw3b[id] = (bf16_t)tw3[id];
    if (id < CC * WW) sw3b[id] = (bf16_t)sw3[id];
    if (id < 2 * WW * CC) {                               // stack sw1 halves
        int r = id >> 8, c = id & 255;                    // sw1 is (128, 512)
        sw1s[id] = (bf16_t)sw1[(size_t)(r & 127) * 512 + (r >> 7) * 256 + c];
    }
}

__global__ void prepx_kernel(const float* __restrict__ x,
                             bf16_t* __restrict__ xc, bf16_t* __restrict__ xT) {
    int id = blockIdx.x * blockDim.x + threadIdx.x;       // B*C*N
    int n = id & (NN - 1), c = (id >> 11) & (CC - 1), b = id >> 19;
    float v = x[id];
    xc[id] = (bf16_t)v;
    xT[((size_t)b * NN + n) * CC + c] = (bf16_t)v;
}

__global__ void xnorm_kernel(const float* __restrict__ x, float* __restrict__ xx) {
    int id = blockIdx.x * blockDim.x + threadIdx.x;       // B*N
    int n = id & (NN - 1), b = id >> 11;
    const float* p = x + (size_t)b * CC * NN + n;
    float s = 0.f;
    for (int c = 0; c < CC; ++c) { float v = p[(size_t)c * NN]; s += v * v; }
    xx[id] = s;
}

// ---------------------------------------------------------------------------
// kNN + fused top-10: one wave per 16-row tile. Gram tile via 8 WMMA (K=256),
// neg_dist = 2*dot - ||xr||^2 - ||xc||^2. All 32 lanes scan: lane owns
// (row = lane&15, col-half = lane>>4), 8 columns/tile into a private top-10;
// half-lists merged once at the end through LDS.
// ---------------------------------------------------------------------------
__global__ void knn_topk_kernel(const bf16_t* __restrict__ xT, const bf16_t* __restrict__ xc,
                                const float* __restrict__ xx, int* __restrict__ idx) {
    __shared__ float sc[8][16][17];     // per-wave 16x16 score tile
    __shared__ float mv[8][32][KK];     // merge staging: values
    __shared__ int   mi[8][32][KK];     // merge staging: indices
    const int lane = threadIdx.x & 31;
    const int w    = threadIdx.x >> 5;
    const int gw   = blockIdx.x * (blockDim.x >> 5) + w;
    const int b    = gw >> 7;                 // 128 row tiles per batch
    const int n0   = (gw & 127) * 16;
    const bf16_t* xTb = xT + (size_t)b * NN * CC;
    const bf16_t* xcb = xc + (size_t)b * CC * NN;
    const float*  xxb = xx + (size_t)b * NN;
    const int hi = lane >> 4, lo = lane & 15;
    const int row = lo, jbase = hi * 8;

    // A fragments for this row tile are loop-invariant: keep in registers.
    v16bf afr[CC / 32];
    #pragma unroll
    for (int kk = 0; kk < CC; kk += 32)
        afr[kk / 32] = load_a_frag(xTb, n0, CC, kk, lane);

    float xxr[8];
    #pragma unroll
    for (int r = 0; r < 8; ++r) xxr[r] = xxb[n0 + hi * 8 + r];

    float tv[KK]; int ti[KK];
    #pragma unroll
    for (int i = 0; i < KK; ++i) { tv[i] = -3.4e38f; ti[i] = 0; }

    for (int ct = 0; ct < NN / 16; ++ct) {
        const int c0 = ct * 16;
        v8f acc = {0.f, 0.f, 0.f, 0.f, 0.f, 0.f, 0.f, 0.f};
        #pragma unroll
        for (int kk = 0; kk < CC; kk += 32) {
            v16bf bf = load_b_frag(xcb, kk, NN, c0, lane);
            acc = wmma_bf16(afr[kk / 32], bf, acc);
        }
        const float xxcv = xxb[c0 + lo];
        #pragma unroll
        for (int r = 0; r < 8; ++r)
            sc[w][hi * 8 + r][lo] = 2.f * acc[r] - xxr[r] - xxcv;
        // cross-lane LDS RAW within the wave: compiler cannot see it
        asm volatile("s_wait_dscnt 0" ::: "memory");
        #pragma unroll
        for (int j = 0; j < 8; ++j)
            topk_insert(sc[w][row][jbase + j], c0 + jbase + j, tv, ti);
    }

    // merge the two half-lists per row (lane l and lane l+16)
    #pragma unroll
    for (int i = 0; i < KK; ++i) { mv[w][lane][i] = tv[i]; mi[w][lane][i] = ti[i]; }
    asm volatile("s_wait_dscnt 0" ::: "memory");
    if (lane < 16) {
        #pragma unroll
        for (int i = 0; i < KK; ++i)
            topk_insert(mv[w][lane + 16][i], mi[w][lane + 16][i], tv, ti);
        int* op = idx + ((size_t)b * NN + n0 + lane) * KK;
        #pragma unroll
        for (int i = 0; i < KK; ++i) op[i] = ti[i];
    }
}

// ---------------------------------------------------------------------------
// Generic batched WMMA GEMM, NT column-tiles per wave (A fragments loaded
// once, reused over 16*NT columns): out[b] = act(W(MxKd) @ X[b] + bias).
// OUTFMT: 0 = f32 (B,M,ncols), 1 = bf16 (B,M,ncols), 2 = bf16 (B,ncols,M)
// ---------------------------------------------------------------------------
template<int MROWS, int KD, bool RELU, bool BIAS, int OUTFMT, int NT>
__global__ void gemm_wmma_kernel(const bf16_t* __restrict__ W, const bf16_t* __restrict__ X,
                                 const float* __restrict__ bias, void* __restrict__ outp,
                                 int ncols) {
    const int lane = threadIdx.x & 31;
    const int gw   = blockIdx.x * (blockDim.x >> 5) + (threadIdx.x >> 5);
    const int ngroups = ncols / (16 * NT), mtiles = MROWS >> 4;
    const int tpb = ngroups * mtiles;
    const int b = gw / tpb;
    if (b >= BB) return;                              // wave-uniform guard
    const int t  = gw % tpb;
    const int mt = t / ngroups, n0 = (t % ngroups) * 16 * NT;
    const bf16_t* Xb = X + (size_t)b * KD * ncols;

    v16bf afr[KD / 32];
    #pragma unroll
    for (int kk = 0; kk < KD; kk += 32)
        afr[kk / 32] = load_a_frag(W, mt * 16, KD, kk, lane);

    v8f acc[NT];
    #pragma unroll
    for (int j = 0; j < NT; ++j) acc[j] = (v8f){0.f, 0.f, 0.f, 0.f, 0.f, 0.f, 0.f, 0.f};

    #pragma unroll
    for (int j = 0; j < NT; ++j) {
        #pragma unroll
        for (int kk = 0; kk < KD; kk += 32) {
            v16bf bf = load_b_frag(Xb, kk, (size_t)ncols, n0 + j * 16, lane);
            acc[j] = wmma_bf16(afr[kk / 32], bf, acc[j]);
        }
    }

    const int hi = lane >> 4, lo = lane & 15;
    #pragma unroll
    for (int j = 0; j < NT; ++j) {
        const int n = n0 + j * 16 + lo;
        if constexpr (OUTFMT == 2) {
            v8bf pack = {};
            #pragma unroll
            for (int r = 0; r < 8; ++r) {
                float v = acc[j][r];
                if (BIAS) v += bias[mt * 16 + hi * 8 + r];
                if (RELU) v = fmaxf(v, 0.f);
                pack[r] = (bf16_t)v;
            }
            bf16_t* o = (bf16_t*)outp + ((size_t)b * ncols + n) * MROWS + mt * 16 + hi * 8;
            *(v8bf*)o = pack;
        } else {
            #pragma unroll
            for (int r = 0; r < 8; ++r) {
                int m = mt * 16 + hi * 8 + r;
                float v = acc[j][r];
                if (BIAS) v += bias[m];
                if (RELU) v = fmaxf(v, 0.f);
                size_t oi = ((size_t)b * MROWS + m) * ncols + n;
                if constexpr (OUTFMT == 0) ((float*)outp)[oi] = v;
                else                       ((bf16_t*)outp)[oi] = (bf16_t)v;
            }
        }
    }
}

// ---------------------------------------------------------------------------
// Grouped temporal conv k=3 (128ch, 32 groups of 4) + bias + relu (tiny).
// ---------------------------------------------------------------------------
__global__ void tconv2_kernel(const bf16_t* __restrict__ t1, const float* __restrict__ tw2,
                              const float* __restrict__ tb2, bf16_t* __restrict__ t2) {
    int id = blockIdx.x * blockDim.x + threadIdx.x;   // B*W*N, (b,ch,n) order
    int n = id & (NN - 1), ch = (id >> 11) & (WW - 1), b = id >> 18;
    int g = ch >> 2;
    const bf16_t* base = t1 + ((size_t)b * WW + g * 4) * NN;
    float a = tb2[ch];
    #pragma unroll
    for (int i = 0; i < 4; ++i)
        #pragma unroll
        for (int d = 0; d < 3; ++d) {
            int nn = n + d - 1;
            if (nn >= 0 && nn < NN)
                a += tw2[(ch * 4 + i) * 3 + d] * (float)base[(size_t)i * NN + nn];
        }
    t2[id] = (bf16_t)fmaxf(a, 0.f);
}

// ---------------------------------------------------------------------------
// Graph fuse: s1 = relu(gather(z1) + z2 + sb1); grouped 1x1 (4x4 per group,
// group == lane, no cross-lane); relu; scatter to wbuf (B,W,K,N) bf16.
// ---------------------------------------------------------------------------
__global__ void graph_fuse_kernel(const bf16_t* __restrict__ zT, const int* __restrict__ idx,
                                  const float* __restrict__ sb1, const float* __restrict__ sw2,
                                  const float* __restrict__ sb2, bf16_t* __restrict__ wbuf) {
    const int lane = threadIdx.x & 31;
    const int gw   = blockIdx.x * (blockDim.x >> 5) + (threadIdx.x >> 5);
    const int b = gw / (NN * KK);
    const int rem = gw % (NN * KK);
    const int n = rem / KK, k = rem % KK;
    const int j = idx[((size_t)b * NN + n) * KK + k];
    const bf16_t* zn = zT + ((size_t)b * NN + n) * (2 * WW);
    const bf16_t* zj = zT + ((size_t)b * NN + j) * (2 * WW);
    const int c0 = lane * 4;
    float v[4];
    #pragma unroll
    for (int i = 0; i < 4; ++i) {
        float a = (float)zj[c0 + i] + (float)zn[WW + c0 + i] + sb1[c0 + i];
        v[i] = fmaxf(a, 0.f);
    }
    #pragma unroll
    for (int o = 0; o < 4; ++o) {
        int ch = c0 + o;
        float a = sb2[ch];
        #pragma unroll
        for (int i = 0; i < 4; ++i) a += sw2[ch * 4 + i] * v[i];
        a = fmaxf(a, 0.f);
        wbuf[(((size_t)b * WW + ch) * KK + k) * NN + n] = (bf16_t)a;
    }
}

// ---------------------------------------------------------------------------
// sw3 GEMM with max-over-k fused in epilogue. sw3 A-fragments hoisted above
// the k loop explicitly. sout[b,m,n] = max_k(sw3 @ w)[m,n,k] + sb3[m].
// ---------------------------------------------------------------------------
__global__ void smax_gemm_kernel(const bf16_t* __restrict__ sw3b, const bf16_t* __restrict__ wbuf,
                                 const float* __restrict__ sb3, float* __restrict__ sout) {
    const int lane = threadIdx.x & 31;
    const int gw   = blockIdx.x * (blockDim.x >> 5) + (threadIdx.x >> 5);
    const int tpb = (CC / 16) * (NN / 16);            // 2048 tiles/batch
    const int b = gw / tpb;
    if (b >= BB) return;
    const int t  = gw % tpb;
    const int mt = t / (NN / 16), nt = t % (NN / 16);

    v16bf afr[WW / 32];
    #pragma unroll
    for (int kk = 0; kk < WW; kk += 32)
        afr[kk / 32] = load_a_frag(sw3b, mt * 16, WW, kk, lane);

    v8f best = {-3.4e38f, -3.4e38f, -3.4e38f, -3.4e38f,
                -3.4e38f, -3.4e38f, -3.4e38f, -3.4e38f};
    for (int k = 0; k < KK; ++k) {
        const bf16_t* Bb = wbuf + (size_t)b * WW * KK * NN + (size_t)k * NN;
        v8f acc = {0.f, 0.f, 0.f, 0.f, 0.f, 0.f, 0.f, 0.f};
        #pragma unroll
        for (int kk = 0; kk < WW; kk += 32) {
            v16bf bf = load_b_frag(Bb, kk, (size_t)KK * NN, nt * 16, lane);
            acc = wmma_bf16(afr[kk / 32], bf, acc);
        }
        #pragma unroll
        for (int r = 0; r < 8; ++r) best[r] = fmaxf(best[r], acc[r]);
    }
    const int hi = lane >> 4, lo = lane & 15;
    #pragma unroll
    for (int r = 0; r < 8; ++r) {
        int m = mt * 16 + hi * 8 + r;
        sout[((size_t)b * CC + m) * NN + nt * 16 + lo] = best[r] + sb3[m];
    }
}

// ---------------------------------------------------------------------------
// Final combine: out = relu(tout + x + sout)
// ---------------------------------------------------------------------------
__global__ void final_kernel(const float* __restrict__ x, const float* __restrict__ tout,
                             const float* __restrict__ sout, float* __restrict__ out) {
    int id = blockIdx.x * blockDim.x + threadIdx.x;   // B*C*N
    out[id] = fmaxf(x[id] + tout[id] + sout[id], 0.f);
}

// ---------------------------------------------------------------------------
// Host launcher
// ---------------------------------------------------------------------------
extern "C" void kernel_launch(void* const* d_in, const int* in_sizes, int n_in,
                              void* d_out, int out_size, void* d_ws, size_t ws_size,
                              hipStream_t stream) {
    const float* x   = (const float*)d_in[0];
    const float* tw1 = (const float*)d_in[1];
    const float* tb1 = (const float*)d_in[2];
    const float* tw2 = (const float*)d_in[3];
    const float* tb2 = (const float*)d_in[4];
    const float* tw3 = (const float*)d_in[5];
    const float* tb3 = (const float*)d_in[6];
    const float* sw1 = (const float*)d_in[7];
    const float* sb1 = (const float*)d_in[8];
    const float* sw2 = (const float*)d_in[9];
    const float* sb2 = (const float*)d_in[10];
    const float* sw3 = (const float*)d_in[11];
    const float* sb3 = (const float*)d_in[12];
    float* out = (float*)d_out;

    char* ws = (char*)d_ws;
    bf16_t* xc16 = (bf16_t*)(ws + OFF_XC16);
    bf16_t* xT16 = (bf16_t*)(ws + OFF_XT16);
    bf16_t* t1   = (bf16_t*)(ws + OFF_T1);
    bf16_t* t2   = (bf16_t*)(ws + OFF_T2);
    bf16_t* zT   = (bf16_t*)(ws + OFF_ZT);
    bf16_t* wbuf = (bf16_t*)(ws + OFF_WBUF);
    float*  tout = (float*) (ws + OFF_TOUT);
    float*  sout = (float*) (ws + OFF_SOUT);
    float*  xx   = (float*) (ws + OFF_XX);
    int*    idx  = (int*)   (ws + OFF_IDX);
    bf16_t* tw1b = (bf16_t*)(ws + OFF_TW1B);
    bf16_t* tw3b = (bf16_t*)(ws + OFF_TW3B);
    bf16_t* sw1s = (bf16_t*)(ws + OFF_SW1S);
    bf16_t* sw3b = (bf16_t*)(ws + OFF_SW3B);

    const int BLK = 256;                      // 8 waves per block (wave32)
    const int wavesPerBlock = BLK / 32;

    // 1) weight conversion + stacking
    convw_kernel<<<(2 * WW * CC + BLK - 1) / BLK, BLK, 0, stream>>>(
        tw1, tw3, sw1, sw3, tw1b, tw3b, sw1s, sw3b);

    // 2) x -> bf16 (both layouts) + norms
    prepx_kernel<<<(BB * CC * NN) / BLK, BLK, 0, stream>>>(x, xc16, xT16);
    xnorm_kernel<<<(BB * NN) / BLK, BLK, 0, stream>>>(x, xx);

    // 3) kNN + top-10 (one wave per 16-row tile)
    knn_topk_kernel<<<(BB * (NN / 16)) / wavesPerBlock, BLK, 0, stream>>>(
        xT16, xc16, xx, idx);

    // 4) temporal: t1 = relu(tw1 @ x + tb1)  (bf16, channel-major)
    gemm_wmma_kernel<WW, CC, true, true, 1, 4>
        <<<(BB * (WW / 16) * (NN / 64)) / wavesPerBlock, BLK, 0, stream>>>(
            tw1b, xc16, tb1, t1, NN);

    // 5) grouped conv k=3 + relu
    tconv2_kernel<<<(BB * WW * NN) / BLK, BLK, 0, stream>>>(t1, tw2, tb2, t2);

    // 6) tout = tw3 @ t2 + tb3  (f32, channel-major)
    gemm_wmma_kernel<CC, WW, false, true, 0, 4>
        <<<(BB * (CC / 16) * (NN / 64)) / wavesPerBlock, BLK, 0, stream>>>(
            tw3b, t2, tb3, tout, NN);

    // 7) z = [sw1_nbr; sw1_ctr] @ x  (bf16, N-major for fast gather)
    gemm_wmma_kernel<2 * WW, CC, false, false, 2, 4>
        <<<(BB * (2 * WW / 16) * (NN / 64)) / wavesPerBlock, BLK, 0, stream>>>(
            sw1s, xc16, nullptr, zT, NN);

    // 8) gather + relu + grouped 1x1 + relu -> wbuf
    graph_fuse_kernel<<<(BB * NN * KK) / wavesPerBlock, BLK, 0, stream>>>(
        zT, idx, sb1, sw2, sb2, wbuf);

    // 9) sout = max_k(sw3 @ wbuf) + sb3 (fused max epilogue)
    smax_gemm_kernel<<<(BB * (CC / 16) * (NN / 16)) / wavesPerBlock, BLK, 0, stream>>>(
        sw3b, wbuf, sb3, sout);

    // 10) out = relu(tout + x + sout)
    final_kernel<<<(BB * CC * NN) / BLK, BLK, 0, stream>>>(x, tout, sout, out);
}
